// CrossFeedingGNNWithAttention_63900523430410
// MI455X (gfx1250) — matrix operand order repaired and checked by reading, not compiled, not run on hardware
//
#include <hip/hip_runtime.h>

typedef __attribute__((ext_vector_type(16))) _Float16 v16h;
typedef __attribute__((ext_vector_type(8)))  _Float16 v8h;
typedef __attribute__((ext_vector_type(2)))  _Float16 v2h;
typedef __attribute__((ext_vector_type(8)))  float    v8f;

#define NEG_SLOPE 0.2f
#define SENC_NEG_INF 0x007FFFFFu   // order-preserving encoding of -inf

__device__ __forceinline__ unsigned fenc(float f) {
  unsigned u = __float_as_uint(f);
  return (u & 0x80000000u) ? ~u : (u | 0x80000000u);
}
__device__ __forceinline__ float fdec(unsigned u) {
  unsigned v = (u & 0x80000000u) ? (u & 0x7FFFFFFFu) : ~u;
  return __uint_as_float(v);
}
__device__ __forceinline__ void wave_lds_fence() {
  asm volatile("s_wait_dscnt 0" ::: "memory");
}
__device__ __forceinline__ v16h cat8(v8h lo, v8h hi) {
  return __builtin_shufflevector(lo, hi, 0, 1, 2, 3, 4, 5, 6, 7,
                                         8, 9, 10, 11, 12, 13, 14, 15);
}

// A fragment for V_WMMA_F32_16X16X32_F16 from a row-major f16 LDS tile:
// lane (m = lane&15, kbase = lane>=16 ? 8 : 0) reads halves
// [kbase .. kbase+7] and [kbase+16 .. kbase+23] of row m  => two b128 loads.
__device__ __forceinline__ v16h afrag_from_lds(const _Float16* row_base,
                                               int rowstride, int m, int kbase,
                                               int koff) {
  const _Float16* p = row_base + m * rowstride + koff + kbase;
  v8h a0 = *(const v8h*)(p);
  v8h a1 = *(const v8h*)(p + 16);
  return cat8(a0, a1);
}

// ---------------------------------------------------------------------------
// Generic Y[M x 64] = act(A[M x KDIM] @ W[KDIM x 64] + bias) via WMMA f16.
// 256 threads = 8 waves; each wave owns one 16-row tile (block covers 128 rows).
// W is staged to LDS pre-swizzled in exact B-fragment order so each lane's
// fragment is one contiguous 32-byte LDS read.
// ---------------------------------------------------------------------------
template<int KDIM, bool HAS_BIAS, bool RELU>
__global__ __launch_bounds__(256) void gemm64_wmma(
    const float* __restrict__ A, const float* __restrict__ W,
    const float* __restrict__ bias, float* __restrict__ Y, int Mrows)
{
  __shared__ __attribute__((aligned(32))) _Float16 wlds[KDIM * 64];
  __shared__ __attribute__((aligned(32))) _Float16 alds[8][16 * 32];

  const int tid  = threadIdx.x;
  const int wave = tid >> 5;
  const int lane = tid & 31;

  // swizzled W staging: wlds[((kc*4+nt)*32+l)*16+h] = W[kc*32+(l>=16?8:0)+(h<8?h:h+8)][nt*16+(l&15)]
  for (int i = tid; i < KDIM * 64; i += 256) {
    int h  = i & 15;
    int l  = (i >> 4) & 31;
    int nt = (i >> 9) & 3;
    int kc = i >> 11;
    int kk = kc * 32 + ((l >= 16) ? 8 : 0) + ((h < 8) ? h : h + 8);
    int col = nt * 16 + (l & 15);
    wlds[i] = (_Float16)W[kk * 64 + col];
  }
  __syncthreads();

  const int row0  = (blockIdx.x * 8 + wave) * 16;
  const int m     = lane & 15;
  const int kbase = (lane >= 16) ? 8 : 0;
  v8f acc[4] = {};

  for (int kc = 0; kc < KDIM / 32; ++kc) {
    // stage 16x32 f32->f16 A tile as packed pairs (coalesced b64 loads)
    for (int i = lane; i < 16 * 16; i += 32) {
      int r = i >> 4, cp = i & 15;
      int gr = row0 + r;
      float2 v = {0.0f, 0.0f};
      if (gr < Mrows) v = *(const float2*)&A[(size_t)gr * KDIM + kc * 32 + 2 * cp];
      v2h hv; hv[0] = (_Float16)v.x; hv[1] = (_Float16)v.y;
      *(v2h*)&alds[wave][r * 32 + 2 * cp] = hv;
    }
    if (kc + 1 < KDIM / 32 && row0 < Mrows)
      __builtin_prefetch(&A[(size_t)row0 * KDIM + (kc + 1) * 32], 0, 1);
    wave_lds_fence();

    v16h afrag = afrag_from_lds(alds[wave], 32, m, kbase, 0);
    #pragma unroll
    for (int nt = 0; nt < 4; ++nt) {
      v16h bfrag = *(const v16h*)&wlds[((kc * 4 + nt) * 32 + lane) * 16];
      acc[nt] = __builtin_amdgcn_wmma_f32_16x16x32_f16(
          false, afrag, false, bfrag, (short)0, acc[nt], false, false);
    }
    wave_lds_fence();   // WAR: fragment reads done before next-iter restage
  }

  const int rbase = (lane >= 16) ? 8 : 0;
  #pragma unroll
  for (int nt = 0; nt < 4; ++nt) {
    const int col = nt * 16 + m;
    float bv = HAS_BIAS ? bias[col] : 0.0f;
    #pragma unroll
    for (int r = 0; r < 8; ++r) {
      int gr = row0 + rbase + r;
      if (gr < Mrows) {
        float v = acc[nt][r] + bv;
        if (RELU) v = fmaxf(v, 0.0f);
        Y[(size_t)gr * 64 + col] = v;
      }
    }
  }
}

// ---------------------------------------------------------------------------
// Fused edge head: hidden = relu([x[src]|x[dst]|e_emb] @ ep1_w + b1) (K=192)
// then edge_pred = hidden @ ep2_w + b2 (N=50, zero-padded to 64), all per-wave.
// ---------------------------------------------------------------------------
__global__ __launch_bounds__(256) void edge_head_wmma(
    const float* __restrict__ x2, const float* __restrict__ e_emb,
    const int* __restrict__ srcs, const int* __restrict__ dsts,
    const float* __restrict__ W1, const float* __restrict__ b1,
    const float* __restrict__ W2, const float* __restrict__ b2,
    float* __restrict__ edge_pred, int E)
{
  __shared__ __attribute__((aligned(32))) _Float16 w1lds[192 * 64];
  __shared__ __attribute__((aligned(32))) _Float16 w2lds[64 * 64];
  __shared__ __attribute__((aligned(32))) _Float16 alds[8][16 * 192];
  __shared__ __attribute__((aligned(32))) _Float16 clds[8][16 * 64];

  const int tid  = threadIdx.x;
  const int wave = tid >> 5;
  const int lane = tid & 31;

  // swizzled B-fragment staging for both weight matrices
  for (int i = tid; i < 192 * 64; i += 256) {
    int h  = i & 15;
    int l  = (i >> 4) & 31;
    int nt = (i >> 9) & 3;
    int kc = i >> 11;
    int kk = kc * 32 + ((l >= 16) ? 8 : 0) + ((h < 8) ? h : h + 8);
    int col = nt * 16 + (l & 15);
    w1lds[i] = (_Float16)W1[kk * 64 + col];
  }
  for (int i = tid; i < 64 * 64; i += 256) {
    int h  = i & 15;
    int l  = (i >> 4) & 31;
    int nt = (i >> 9) & 3;
    int kc = i >> 11;
    int kk = kc * 32 + ((l >= 16) ? 8 : 0) + ((h < 8) ? h : h + 8);
    int col = nt * 16 + (l & 15);
    w2lds[i] = (_Float16)((col < 50) ? W2[kk * 50 + col] : 0.0f);   // zero-pad N
  }
  __syncthreads();

  const int e0    = (blockIdx.x * 8 + wave) * 16;
  const int m     = lane & 15;
  const int kbase = (lane >= 16) ? 8 : 0;
  const int rbase = (lane >= 16) ? 8 : 0;

  // gather-concat 16x192 tile: [x2[src] | x2[dst] | e_emb], packed-pair stores
  for (int i = lane; i < 16 * 96; i += 32) {
    int r = i / 96, c = (i % 96) * 2;
    int e = e0 + r;
    float2 v = {0.0f, 0.0f};
    if (e < E) {
      if (c < 64)       v = *(const float2*)&x2[(size_t)srcs[e] * 64 + c];
      else if (c < 128) v = *(const float2*)&x2[(size_t)dsts[e] * 64 + (c - 64)];
      else              v = *(const float2*)&e_emb[(size_t)e * 64 + (c - 128)];
    }
    v2h hv; hv[0] = (_Float16)v.x; hv[1] = (_Float16)v.y;
    *(v2h*)&alds[wave][r * 192 + c] = hv;
  }
  wave_lds_fence();

  // stage 1: K = 192
  v8f acc[4] = {};
  for (int kc = 0; kc < 6; ++kc) {
    v16h afrag = afrag_from_lds(alds[wave], 192, m, kbase, kc * 32);
    #pragma unroll
    for (int nt = 0; nt < 4; ++nt) {
      v16h bfrag = *(const v16h*)&w1lds[((kc * 4 + nt) * 32 + lane) * 16];
      acc[nt] = __builtin_amdgcn_wmma_f32_16x16x32_f16(
          false, afrag, false, bfrag, (short)0, acc[nt], false, false);
    }
  }

  // relu + bias1, C-layout -> row-major LDS (f16), re-fed as A-layout below
  #pragma unroll
  for (int nt = 0; nt < 4; ++nt) {
    const int col = nt * 16 + m;
    float bv = b1[col];
    #pragma unroll
    for (int r = 0; r < 8; ++r) {
      float v = fmaxf(acc[nt][r] + bv, 0.0f);
      clds[wave][(rbase + r) * 64 + col] = (_Float16)v;
    }
  }
  wave_lds_fence();

  // stage 2: K = 64 against padded ep2_w
  v8f acc2[4] = {};
  #pragma unroll
  for (int kc = 0; kc < 2; ++kc) {
    v16h afrag = afrag_from_lds(clds[wave], 64, m, kbase, kc * 32);
    #pragma unroll
    for (int nt = 0; nt < 4; ++nt) {
      v16h bfrag = *(const v16h*)&w2lds[((kc * 4 + nt) * 32 + lane) * 16];
      acc2[nt] = __builtin_amdgcn_wmma_f32_16x16x32_f16(
          false, afrag, false, bfrag, (short)0, acc2[nt], false, false);
    }
  }

  #pragma unroll
  for (int nt = 0; nt < 4; ++nt) {
    const int col = nt * 16 + m;
    if (col < 50) {
      float bv = b2[col];
      #pragma unroll
      for (int r = 0; r < 8; ++r) {
        int e = e0 + rbase + r;
        if (e < E) edge_pred[(size_t)e * 50 + col] = acc2[nt][r] + bv;
      }
    }
  }
}

// ---------------------------------------------------------------------------
// Attention / elementwise helper kernels (f32 scalar path)
// ---------------------------------------------------------------------------
__global__ void weff_kernel(const float* __restrict__ W,
                            const float* __restrict__ att, float* weff) {
  int k = threadIdx.x;  // 64 threads
  float s = 0.0f;
  #pragma unroll
  for (int j = 0; j < 64; ++j) s += W[k * 64 + j] * att[j];
  weff[k] = s;
}

__global__ void att_dots_kernel(const float* __restrict__ h,
                                const float* __restrict__ att_src,
                                const float* __restrict__ att_dst,
                                float* __restrict__ a_s, float* __restrict__ a_d, int N) {
  int n = blockIdx.x * blockDim.x + threadIdx.x;
  if (n >= N) return;
  const float4* row = (const float4*)(h + (size_t)n * 64);
  const float4* as4 = (const float4*)att_src;
  const float4* ad4 = (const float4*)att_dst;
  float s = 0.0f, d = 0.0f;
  #pragma unroll
  for (int i = 0; i < 16; ++i) {
    float4 r = row[i], a = as4[i], b = ad4[i];
    s += r.x * a.x + r.y * a.y + r.z * a.z + r.w * a.w;
    d += r.x * b.x + r.y * b.y + r.z * b.z + r.w * b.w;
  }
  a_s[n] = s; a_d[n] = d;
}

__global__ void init_senc_kernel(unsigned* __restrict__ senc, int N) {
  int n = blockIdx.x * blockDim.x + threadIdx.x;
  if (n < N) senc[n] = SENC_NEG_INF;
}

__global__ void edge_logits_kernel(const float* __restrict__ e_emb,
                                   const float* __restrict__ weff,
                                   const float* __restrict__ a_s,
                                   const float* __restrict__ a_d,
                                   const int* __restrict__ srcs,
                                   const int* __restrict__ dsts,
                                   float* __restrict__ logits,
                                   unsigned* __restrict__ senc, int E) {
  int e = blockIdx.x * blockDim.x + threadIdx.x;
  if (e >= E) return;
  const float4* row = (const float4*)(e_emb + (size_t)e * 64);
  const float4* w4  = (const float4*)weff;
  float ae = 0.0f;
  #pragma unroll
  for (int i = 0; i < 16; ++i) {
    float4 r = row[i], w = w4[i];
    ae += r.x * w.x + r.y * w.y + r.z * w.z + r.w * w.w;
  }
  int d = dsts[e];
  float l = a_s[srcs[e]] + a_d[d] + ae;
  l = (l >= 0.0f) ? l : NEG_SLOPE * l;
  logits[e] = l;
  atomicMax(&senc[d], fenc(l));
}

__global__ void segmax_fix_kernel(const unsigned* __restrict__ senc,
                                  float* __restrict__ smax, int N) {
  int n = blockIdx.x * blockDim.x + threadIdx.x;
  if (n >= N) return;
  float v = fdec(senc[n]);
  smax[n] = isfinite(v) ? v : 0.0f;
}

__global__ void exp_sum_kernel(const float* __restrict__ logits,
                               const float* __restrict__ smax,
                               const int* __restrict__ dsts,
                               float* __restrict__ exbuf,
                               float* __restrict__ denom, int E) {
  int e = blockIdx.x * blockDim.x + threadIdx.x;
  if (e >= E) return;
  int d = dsts[e];
  float ex = expf(logits[e] - smax[d]);
  exbuf[e] = ex;
  atomicAdd(&denom[d], ex);
}

// one block (64 threads = channels) per edge: alpha + scatter-add aggregation
__global__ void alpha_agg_kernel(const float* __restrict__ exbuf,
                                 const float* __restrict__ denom,
                                 const int* __restrict__ srcs,
                                 const int* __restrict__ dsts,
                                 const float* __restrict__ h,
                                 float* __restrict__ alpha_out,
                                 float* __restrict__ agg) {
  int e = blockIdx.x;
  int c = threadIdx.x;
  int s = srcs[e], d = dsts[e];
  float a = exbuf[e] / (denom[d] + 1e-16f);
  if (c == 0) alpha_out[e] = a;
  atomicAdd(&agg[(size_t)d * 64 + c], a * h[(size_t)s * 64 + c]);
}

__global__ void bias_relu_kernel(const float* __restrict__ agg,
                                 const float* __restrict__ bias,
                                 float* __restrict__ xout, int total) {
  int i = blockIdx.x * blockDim.x + threadIdx.x;
  if (i >= total) return;
  xout[i] = fmaxf(agg[i] + bias[i & 63], 0.0f);
}

__global__ void node_head2_kernel(const float* __restrict__ t,
                                  const float* __restrict__ np2_w,
                                  const float* __restrict__ np2_b,
                                  float* __restrict__ node_pred, int N) {
  int n = blockIdx.x * blockDim.x + threadIdx.x;
  if (n >= N) return;
  const float4* row = (const float4*)(t + (size_t)n * 64);
  const float4* w4  = (const float4*)np2_w;
  float s = 0.0f;
  #pragma unroll
  for (int i = 0; i < 16; ++i) {
    float4 r = row[i], w = w4[i];
    s += r.x * w.x + r.y * w.y + r.z * w.z + r.w * w.w;
  }
  node_pred[n] = s + np2_b[0];
}

// ---------------------------------------------------------------------------
extern "C" void kernel_launch(void* const* d_in, const int* in_sizes, int n_in,
                              void* d_out, int out_size, void* d_ws, size_t ws_size,
                              hipStream_t stream) {
  (void)n_in; (void)out_size; (void)ws_size;

  const float* node_features = (const float*)d_in[0];
  const float* edge_attr     = (const float*)d_in[1];
  const int*   edge_index    = (const int*)d_in[2];
  const float* node_fc_w = (const float*)d_in[3];
  const float* node_fc_b = (const float*)d_in[4];
  const float* edge_fc_w = (const float*)d_in[5];
  const float* edge_fc_b = (const float*)d_in[6];
  const float* g_lin_w[2]    = {(const float*)d_in[7],  (const float*)d_in[13]};
  const float* g_att_src[2]  = {(const float*)d_in[8],  (const float*)d_in[14]};
  const float* g_att_dst[2]  = {(const float*)d_in[9],  (const float*)d_in[15]};
  const float* g_att_edge[2] = {(const float*)d_in[10], (const float*)d_in[16]};
  const float* g_lin_edge[2] = {(const float*)d_in[11], (const float*)d_in[17]};
  const float* g_bias[2]     = {(const float*)d_in[12], (const float*)d_in[18]};
  const float* np1_w = (const float*)d_in[19];
  const float* np1_b = (const float*)d_in[20];
  const float* np2_w = (const float*)d_in[21];
  const float* np2_b = (const float*)d_in[22];
  const float* ep1_w = (const float*)d_in[23];
  const float* ep1_b = (const float*)d_in[24];
  const float* ep2_w = (const float*)d_in[25];
  const float* ep2_b = (const float*)d_in[26];

  const int N = in_sizes[0] / 64;
  const int E = in_sizes[1] / 32;
  const int* srcs = edge_index;
  const int* dsts = edge_index + E;

  float* out = (float*)d_out;
  float* node_pred = out;                              // [N]
  float* edge_pred = out + N;                          // [E,50]
  float* alpha[2];
  alpha[0] = out + N + (size_t)E * 50;                 // [E]
  alpha[1] = alpha[0] + E;                             // [E]

  float* WS = (float*)d_ws;
  size_t off = 0;
  float* e_emb = WS + off; off += (size_t)E * 64;
  float* bufA  = WS + off; off += (size_t)N * 64;      // x0, later agg (both layers)
  float* bufB  = WS + off; off += (size_t)N * 64;      // h1 / h2 / t
  float* x1    = WS + off; off += (size_t)N * 64;
  float* x2    = WS + off; off += (size_t)N * 64;
  float* a_s   = WS + off; off += N;
  float* a_d   = WS + off; off += N;
  float* smax  = WS + off; off += N;
  float* denom = WS + off; off += N;
  unsigned* senc = (unsigned*)(WS + off); off += N;
  float* weff  = WS + off; off += 64;
  float* logits = WS + off; off += E;
  float* exbuf  = WS + off; off += E;

  const int gN128 = (N + 127) / 128;
  const int gE128 = (E + 127) / 128;
  const int gN256 = (N + 255) / 256;
  const int gE256 = (E + 255) / 256;

  // x0 = node_features @ node_fc_w + b ;  e_emb = edge_attr @ edge_fc_w + b
  gemm64_wmma<64, true, false><<<gN128, 256, 0, stream>>>(node_features, node_fc_w, node_fc_b, bufA, N);
  gemm64_wmma<32, true, false><<<gE128, 256, 0, stream>>>(edge_attr, edge_fc_w, edge_fc_b, e_emb, E);

  const float* xin = bufA;
  float* xouts[2] = {x1, x2};
  for (int L = 0; L < 2; ++L) {
    float* h    = bufB;
    float* xout = xouts[L];
    // h = xin @ lin_w ; weff = lin_edge_w @ att_edge (so a_e = e_emb . weff)
    gemm64_wmma<64, false, false><<<gN128, 256, 0, stream>>>(xin, g_lin_w[L], nullptr, h, N);
    weff_kernel<<<1, 64, 0, stream>>>(g_lin_edge[L], g_att_edge[L], weff);
    att_dots_kernel<<<gN256, 256, 0, stream>>>(h, g_att_src[L], g_att_dst[L], a_s, a_d, N);

    init_senc_kernel<<<gN256, 256, 0, stream>>>(senc, N);
    hipMemsetAsync(denom, 0, (size_t)N * sizeof(float), stream);
    hipMemsetAsync(bufA, 0, (size_t)N * 64 * sizeof(float), stream);  // agg

    edge_logits_kernel<<<gE256, 256, 0, stream>>>(e_emb, weff, a_s, a_d, srcs, dsts, logits, senc, E);
    segmax_fix_kernel<<<gN256, 256, 0, stream>>>(senc, smax, N);
    exp_sum_kernel<<<gE256, 256, 0, stream>>>(logits, smax, dsts, exbuf, denom, E);
    alpha_agg_kernel<<<E, 64, 0, stream>>>(exbuf, denom, srcs, dsts, h, alpha[L], bufA);
    bias_relu_kernel<<<(N * 64 + 255) / 256, 256, 0, stream>>>(bufA, g_bias[L], xout, N * 64);
    xin = xout;
  }

  // node head: t = relu(x2 @ np1_w + b) ; node_pred = t @ np2_w + b
  gemm64_wmma<64, true, true><<<gN128, 256, 0, stream>>>(x2, np1_w, np1_b, bufB, N);
  node_head2_kernel<<<gN256, 256, 0, stream>>>(bufB, np2_w, np2_b, node_pred, N);

  // fused edge head (gather -> ep1 -> relu -> ep2)
  edge_head_wmma<<<gE128, 256, 0, stream>>>(x2, e_emb, srcs, dsts,
                                            ep1_w, ep1_b, ep2_w, ep2_b, edge_pred, E);
}
